// MediumTensorVM_4810363372677
// MI455X (gfx1250) — matrix operand order repaired and checked by reading, not compiled, not run on hardware
//
#include <hip/hip_runtime.h>
#include <math.h>

typedef __attribute__((ext_vector_type(16))) _Float16 v16h;
typedef __attribute__((ext_vector_type(8)))  _Float16 v8h;
typedef __attribute__((ext_vector_type(8)))  float    v8f;

#define WMMA_F16(a, b, c) \
  __builtin_amdgcn_wmma_f32_16x16x32_f16(false, (a), false, (b), (short)0, (c), false, false)

#define NRES 300
#define PLANE_HW (NRES * NRES)
#define C_TOT 32
#define A_COMP 24
#define APP_DIM 27
#define IN_K 66      // real K of layer 1
#define K1 96        // padded K of layer 1 (3 x 32)
#define K1P 104      // LDS leading dim for A tile (bank-conflict padding, 16B aligned)
#define FEAT 128
#define FEATP 136    // LDS leading dim for hidden tile
#define TM 64        // samples per block
#define THREADS 128  // 4 waves

#define CAT16(lo, hi) __builtin_shufflevector((lo), (hi), 0,1,2,3,4,5,6,7,8,9,10,11,12,13,14,15)

// A: 16x32 (MxK) fragment from row-major [row][k] (leading dim ld halves).
// Contiguous K order: lanes 0-15 hold K {0..15}, lanes 16-31 hold K {16..31}.
// => two contiguous 8-half chunks per lane => 2 x ds_load_b128.
__device__ inline v16h load_a_frag(const _Float16* A, int row0, int k0, int ld) {
  const int lane = threadIdx.x & 31;
  const _Float16* p = A + (row0 + (lane & 15)) * ld + k0 + ((lane >> 4) << 4);
  const v8h lo = *(const v8h*)(p);
  const v8h hi = *(const v8h*)(p + 8);
  return CAT16(lo, hi);
}

// B: 32x16 (KxN) fragment from pre-swizzled LDS:
// layout [tile][chunk(2)][lane(32)][8 halves] => 2 x ds_load_b128, contiguous per lane.
__device__ inline v16h load_b_sw(const _Float16* B, int tile) {
  const int lane = threadIdx.x & 31;
  const _Float16* p = B + (size_t)tile * 512 + lane * 8;
  const v8h lo = *(const v8h*)(p);
  const v8h hi = *(const v8h*)(p + 256);
  return CAT16(lo, hi);
}

__device__ inline float softplusf(float x) {
  return (x > 20.0f) ? x : log1pf(expf(x));
}
__device__ inline float sigmoidf(float x) { return 1.0f / (1.0f + expf(x)); }

__global__ __launch_bounds__(THREADS)
void medium_tensorvm_fused(const float* __restrict__ xyz,
                           const float* __restrict__ vdirs,
                           const float* __restrict__ plane,
                           const float* __restrict__ line,
                           const float* __restrict__ basis,
                           const float* __restrict__ W1, const float* __restrict__ b1,
                           const float* __restrict__ W2, const float* __restrict__ b2,
                           const float* __restrict__ W3, const float* __restrict__ b3,
                           float* __restrict__ out, int n) {
  // LDS: 63488 B total
  __shared__ __align__(16) unsigned char smem[63488];
  _Float16* sA     = (_Float16*)smem;                    // 64 x 104 f16 = 13312 B
  float*    sBasis = (float*)(smem + 13312);             // 27 x 72  f32 =  7776 B (in W region)
  _Float16* sW1    = (_Float16*)(smem + 13312 + 8192);   // swizzled W1: 24 tiles*512 = 24576 B
  _Float16* sW     = (_Float16*)(smem + 13312);          // reused: swizzled W2 (32768 B) / W3
  _Float16* sH     = (_Float16*)(smem + 46080);          // 64 x 136 f16 = 17408 B

  const int tid  = threadIdx.x;
  const int lane = tid & 31;
  const int wv   = tid >> 5;            // wave id 0..3 owns M rows [16*wv, 16*wv+16)
  const int colL = lane & 15;
  const int rowAdd = (lane >> 4) << 3;  // lanes 16..31 -> +8 rows in C/D layout
  const long long m0 = (long long)blockIdx.x * TM;

  float* outRGB  = out;
  float* outBS   = out + (long long)3 * n;
  float* outATTN = out + (long long)6 * n;
  float* outDEN  = out + (long long)9 * n;

  // ---------- Phase 0: stage basis (f32) + W1 pre-swizzled into B-fragment order ----------
  for (int i = tid; i < APP_DIM * 72; i += THREADS) sBasis[i] = basis[i];
  // addr = tile*512 + chunk*256 + l*8 + j ; tile = kt*8+nt ; k = kt*32 + 16*(l>=16) + chunk*8 + j
  for (int i = tid; i < 24 * 512; i += THREADS) {
    const int j = i & 7, l = (i >> 3) & 31, chunk = (i >> 8) & 1, tile = i >> 9;
    const int kt = tile >> 3, nt = tile & 7;
    const int k = kt * 32 + ((l >> 4) << 4) + chunk * 8 + j;
    const int nn = nt * 16 + (l & 15);
    sW1[i] = (k < IN_K) ? (_Float16)W1[nn * IN_K + k] : (_Float16)0.0f;
  }
  __syncthreads();

  // ---------- Phase 1: per-sample features -> sA, density -> out ----------
  if (tid < TM) {
    const long long s = m0 + tid;
    if (s < n) {
      float g[3], dv[3];
#pragma unroll
      for (int i = 0; i < 3; ++i) {
        g[i]  = xyz[s * 3 + i];    // xyz_n == xyz for these constants
        dv[i] = vdirs[s * 3 + i];
      }
      const int MA[3] = {0, 0, 1}, MB[3] = {1, 2, 2}, VM[3] = {2, 1, 0};
      float appf[APP_DIM];
#pragma unroll
      for (int j = 0; j < APP_DIM; ++j) appf[j] = 0.0f;
      float dens = 0.0f;

      for (int pl = 0; pl < 3; ++pl) {
        const float ix = (g[MA[pl]] + 1.0f) * 0.5f * (NRES - 1);
        const float iy = (g[MB[pl]] + 1.0f) * 0.5f * (NRES - 1);
        float fx = fminf(fmaxf(floorf(ix), 0.0f), (float)(NRES - 2));
        float fy = fminf(fmaxf(floorf(iy), 0.0f), (float)(NRES - 2));
        const int x0 = (int)fx, y0 = (int)fy;
        const float wx = ix - fx, wy = iy - fy;
        const float il = (g[VM[pl]] + 1.0f) * 0.5f * (NRES - 1);
        float fl = fminf(fmaxf(floorf(il), 0.0f), (float)(NRES - 2));
        const int l0 = (int)fl;
        const float wl = il - fl;

        const float* Pb = plane + (long long)pl * C_TOT * PLANE_HW + y0 * NRES + x0;
        const float* Lb = line  + (long long)pl * C_TOT * NRES + l0;
        for (int c = 0; c < C_TOT; ++c) {
          const float* P = Pb + (long long)c * PLANE_HW;
          const float v00 = P[0], v01 = P[1], v10 = P[NRES], v11 = P[NRES + 1];
          const float pf = (v00 * (1.0f - wy) + v10 * wy) * (1.0f - wx) +
                           (v01 * (1.0f - wy) + v11 * wy) * wx;
          const float* L = Lb + c * NRES;
          const float lf = L[0] * (1.0f - wl) + L[1] * wl;
          const float prod = pf * lf;
          if (c < A_COMP) {
            const int kIdx = pl * A_COMP + c;
#pragma unroll
            for (int j = 0; j < APP_DIM; ++j)
              appf[j] += sBasis[j * 72 + kIdx] * prod;
          } else {
            dens += prod;
          }
        }
      }
      outDEN[s] = dens;

      _Float16* row = sA + tid * K1P;
#pragma unroll
      for (int j = 0; j < APP_DIM; ++j) row[j] = (_Float16)appf[j];
#pragma unroll
      for (int i = 0; i < 3; ++i) row[APP_DIM + i] = (_Float16)dv[i];
#pragma unroll
      for (int d = 0; d < 3; ++d)
#pragma unroll
        for (int f = 0; f < 6; ++f) {
          const float ang = dv[d] * (float)(1 << f);
          row[30 + d * 6 + f] = (_Float16)sinf(ang);
          row[48 + d * 6 + f] = (_Float16)cosf(ang);
        }
#pragma unroll
      for (int k = IN_K; k < K1; ++k) row[k] = (_Float16)0.0f;
    }
  }
  __syncthreads();

  const int row0 = wv * 16;

  // ---------- Phase 2: layer 1 — (64x96) x (96x128), bias+ReLU -> sH ----------
  {
    v8f acc[8] = {};
    for (int kt = 0; kt < 3; ++kt) {
      const v16h a = load_a_frag(sA, row0, kt * 32, K1P);
#pragma unroll
      for (int nt = 0; nt < 8; ++nt) {
        const v16h b = load_b_sw(sW1, kt * 8 + nt);
        acc[nt] = WMMA_F16(a, b, acc[nt]);
      }
    }
#pragma unroll
    for (int nt = 0; nt < 8; ++nt) {
      const int nn = nt * 16 + colL;
      const float bias = b1[nn];
#pragma unroll
      for (int r = 0; r < 8; ++r) {
        float h = acc[nt][r] + bias;
        h = fmaxf(h, 0.0f);
        sH[(row0 + rowAdd + r) * FEATP + nn] = (_Float16)h;
      }
    }
  }
  __syncthreads();

  // ---------- Phase 3: stage W2 pre-swizzled (32 tiles) ----------
  for (int i = tid; i < 32 * 512; i += THREADS) {
    const int j = i & 7, l = (i >> 3) & 31, chunk = (i >> 8) & 1, tile = i >> 9;
    const int kt = tile >> 3, nt = tile & 7;
    const int k = kt * 32 + ((l >> 4) << 4) + chunk * 8 + j;
    const int nn = nt * 16 + (l & 15);
    sW[i] = (_Float16)W2[nn * FEAT + k];
  }
  __syncthreads();

  // ---------- Phase 4: layer 2 — (64x128) x (128x128), bias+ReLU, in-place per-wave ----------
  {
    v8f acc[8] = {};
    for (int kt = 0; kt < 4; ++kt) {
      const v16h a = load_a_frag(sH, row0, kt * 32, FEATP);
#pragma unroll
      for (int nt = 0; nt < 8; ++nt) {
        const v16h b = load_b_sw(sW, kt * 8 + nt);
        acc[nt] = WMMA_F16(a, b, acc[nt]);
      }
    }
#pragma unroll
    for (int nt = 0; nt < 8; ++nt) {
      const int nn = nt * 16 + colL;
      const float bias = b2[nn];
#pragma unroll
      for (int r = 0; r < 8; ++r) {
        float h = acc[nt][r] + bias;
        h = fmaxf(h, 0.0f);
        sH[(row0 + rowAdd + r) * FEATP + nn] = (_Float16)h;  // own rows only: safe
      }
    }
  }
  __syncthreads();

  // ---------- Phase 5: stage W3 pre-swizzled (4 tiles, cols >= 9 zero) ----------
  for (int i = tid; i < 4 * 512; i += THREADS) {
    const int j = i & 7, l = (i >> 3) & 31, chunk = (i >> 8) & 1, kt = i >> 9;
    const int k = kt * 32 + ((l >> 4) << 4) + chunk * 8 + j;
    const int nn = l & 15;
    sW[i] = (nn < 9) ? (_Float16)W3[nn * FEAT + k] : (_Float16)0.0f;
  }
  __syncthreads();

  // ---------- Phase 6: layer 3 — (64x128) x (128x16) + activations ----------
  {
    v8f acc = {};
    for (int kt = 0; kt < 4; ++kt) {
      const v16h a = load_a_frag(sH, row0, kt * 32, FEATP);
      const v16h b = load_b_sw(sW, kt);
      acc = WMMA_F16(a, b, acc);
    }
    const int nn = colL;  // output channel 0..15 (>=9 unused)
    const float bias = (nn < 9) ? b3[nn] : 0.0f;
#pragma unroll
    for (int r = 0; r < 8; ++r) {
      const long long s = m0 + row0 + rowAdd + r;
      if (s < n && nn < 9) {
        const float v = acc[r] + bias;
        if (nn < 3)      outRGB[s * 3 + nn]        = sigmoidf(v);
        else if (nn < 6) outBS[s * 3 + (nn - 3)]   = softplusf(v);
        else             outATTN[s * 3 + (nn - 6)] = softplusf(v);
      }
    }
  }
}

extern "C" void kernel_launch(void* const* d_in, const int* in_sizes, int n_in,
                              void* d_out, int out_size, void* d_ws, size_t ws_size,
                              hipStream_t stream) {
  const float* xyz   = (const float*)d_in[0];
  const float* vdirs = (const float*)d_in[1];
  const float* plane = (const float*)d_in[2];
  const float* line  = (const float*)d_in[3];
  const float* basis = (const float*)d_in[4];
  const float* W1    = (const float*)d_in[5];
  const float* b1    = (const float*)d_in[6];
  const float* W2    = (const float*)d_in[7];
  const float* b2    = (const float*)d_in[8];
  const float* W3    = (const float*)d_in[9];
  const float* b3    = (const float*)d_in[10];
  float* out = (float*)d_out;
  const int n = in_sizes[0] / 3;
  const int blocks = (n + TM - 1) / TM;
  medium_tensorvm_fused<<<blocks, THREADS, 0, stream>>>(
      xyz, vdirs, plane, line, basis, W1, b1, W2, b2, W3, b3, out, n);
  (void)d_ws; (void)ws_size; (void)n_in; (void)out_size;
}
